// ScoreNet_21096879358619
// MI455X (gfx1250) — matrix operand order repaired and checked by reading, not compiled, not run on hardware
//
#include <hip/hip_runtime.h>

typedef _Float16 half_t;
typedef __attribute__((ext_vector_type(16))) _Float16 v16h;
typedef __attribute__((ext_vector_type(8)))  _Float16 v8h;
typedef __attribute__((ext_vector_type(2)))  _Float16 v2h;
typedef __attribute__((ext_vector_type(8)))  float    v8f;
typedef __attribute__((ext_vector_type(4)))  float    v4f;

#define PB    64          // batch
#define PP    55          // particles
#define NLAY  6
#define NNODE (PB*PP)     // 3520
#define EPB   (PP*(PP-1)) // 2970
#define NEDGE (PB*EPB)    // 190080
#define DEG   (PP-1)      // 54

#define AS  40            // LDS A-tile row stride (halves), mult of 8 for b128
#define BTS 40            // LDS B-tile row stride (halves), mult of 8 for b128
#define A_SZ (64 * AS)    // halves per A buffer
#define B_SZ (256 * BTS)  // halves per B buffer

__device__ __forceinline__ float siluf(float v) { return v / (1.0f + __expf(-v)); }
__device__ __forceinline__ float sigmf(float v) { return 1.0f / (1.0f + __expf(-v)); }

__device__ __forceinline__ v8f wmma16(const v16h a, const v16h b, const v8f c) {
  return __builtin_amdgcn_wmma_f32_16x16x32_f16(false, a, false, b, (short)0, c, false, false);
}

// Async global<->LDS 16B/lane (ISA 15.18.3), tracked by ASYNCcnt.
__device__ __forceinline__ void async_copy_b128(void* ldsDst, const void* gSrc) {
  unsigned lds = (unsigned)(unsigned long long)ldsDst;  // low 32 bits = LDS offset
  asm volatile("global_load_async_to_lds_b128 %0, %1, off"
               :: "v"(lds), "v"(gSrc) : "memory");
}
__device__ __forceinline__ void async_store_b128(void* gDst, const void* ldsSrc) {
  unsigned lds = (unsigned)(unsigned long long)ldsSrc;
  asm volatile("global_store_async_from_lds_b128 %0, %1, off"
               :: "v"(gDst), "v"(lds) : "memory");
}
__device__ __forceinline__ void wait_async() {
  asm volatile("s_wait_asynccnt 0x0" ::: "memory");
}

// A fragment: 16x32 f16 (ISA 7.12.2): lanes 0-15 hold k {0..7,16..23}, 16-31 {8..15,24..31}
__device__ __forceinline__ v16h load_afrag(const half_t* aT, int mgrp, int lane) {
  int m   = mgrp * 16 + (lane & 15);
  int khb = (lane >> 4) & 1;
  const half_t* p = aT + m * AS + khb * 8;
  v8h lo = *(const v8h*)(p);
  v8h hi = *(const v8h*)(p + 16);
  v16h r;
#pragma unroll
  for (int i = 0; i < 8; i++) { r[i] = lo[i]; r[8 + i] = hi[i]; }
  return r;
}

// B fragment: 32x16 f16: lanes 0-15 K {0..7,16..23}, lanes 16-31 {8..15,24..31}; bTt is k-contiguous.
__device__ __forceinline__ v16h load_bfrag(const half_t* bTt, int ncol0, int lane) {
  int n  = ncol0 + (lane & 15);
  int kh = ((lane >> 4) & 1) * 8;
  const half_t* p = bTt + n * BTS + kh;
  v8h lo = *(const v8h*)(p);
  v8h hi = *(const v8h*)(p + 16);
  v16h r;
#pragma unroll
  for (int i = 0; i < 8; i++) { r[i] = lo[i]; r[8 + i] = hi[i]; }
  return r;
}

// Stage WT[n][kk..kk+31] -> bTt (WT pre-transposed: [256][K], k-contiguous).
__device__ __forceinline__ void stage_wtile(half_t* bTt, const half_t* WT, int K,
                                            int kk, int tid) {
#pragma unroll
  for (int i = 0; i < 4; i++) {
    int q  = tid + i * 256;
    int n  = q >> 2;
    int kq = (q & 3) * 8;
    async_copy_b128(bTt + n * BTS + kq, WT + (long)n * K + kk + kq);
  }
}

__device__ __forceinline__ void stage_atile(half_t* aT, const half_t* A, long rowBase,
                                            int lda, int kk, int tid) {
  int m  = tid >> 2;
  int kq = (tid & 3) * 8;
  async_copy_b128(aT + m * AS + kq, A + (rowBase + m) * (long)lda + kk + kq);
}

// ---------------------------------------------------------------------------
// Generic GEMM: out[M,256] = act(A[M,K]f16 @ WT[256,K]f16 + bias)
// MODE: 1=silu, 2=residual add into outF32 (mirrored), 4=write f16, 8=f32 write
// Double-buffered async LDS staging: one barrier per K-step.
// ---------------------------------------------------------------------------
template <int MODE>
__global__ __launch_bounds__(256)
void gemm_kernel(const half_t* __restrict__ A, int lda, int K,
                 const half_t* __restrict__ WT, const float* __restrict__ bias,
                 float* __restrict__ outF32, half_t* __restrict__ outF16) {
  __shared__ half_t aT[2][A_SZ];
  __shared__ half_t bTt[2][B_SZ];
  int  tid  = threadIdx.x;
  int  lane = tid & 31, wave = tid >> 5;
  int  mgrp = wave >> 1, ngrp = wave & 1;
  long rowBase = (long)blockIdx.x * 64;

  v8f acc[8];
#pragma unroll
  for (int t = 0; t < 8; t++)
#pragma unroll
    for (int r = 0; r < 8; r++) acc[t][r] = 0.0f;

  stage_atile(aT[0], A, rowBase, lda, 0, tid);
  stage_wtile(bTt[0], WT, K, 0, tid);
  wait_async();
  __syncthreads();

  int cur = 0;
  for (int kk = 0; kk < K; kk += 32) {
    int nxt = cur ^ 1;
    if (kk + 32 < K) {                 // prefetch next chunk under the WMMAs
      stage_atile(aT[nxt], A, rowBase, lda, kk + 32, tid);
      stage_wtile(bTt[nxt], WT, K, kk + 32, tid);
    }
    v16h af = load_afrag(aT[cur], mgrp, lane);
#pragma unroll
    for (int t = 0; t < 8; t++) {
      v16h bf = load_bfrag(bTt[cur], ngrp * 128 + t * 16, lane);
      acc[t]  = wmma16(af, bf, acc[t]);
    }
    wait_async();
    __syncthreads();
    cur = nxt;
  }

#pragma unroll
  for (int t = 0; t < 8; t++) {
    int   col = ngrp * 128 + t * 16 + (lane & 15);
    float bv  = bias ? bias[col] : 0.0f;
#pragma unroll
    for (int r = 0; r < 8; r++) {
      long  row = rowBase + mgrp * 16 + ((lane >> 4) & 1) * 8 + r;
      float v   = acc[t][r] + bv;
      if (MODE & 1) v = siluf(v);
      long idx = row * 256 + col;
      if (MODE & 2) { float o = outF32[idx] + v; outF32[idx] = o; v = o; }
      else if (MODE & 8) outF32[idx] = v;
      if (MODE & 4) outF16[idx] = (half_t)v;
    }
  }
}

// ---------------------------------------------------------------------------
// Fused edge MLP: A row built in LDS as silu(Hh[r]+Hc[c]+rad*w1r+ea*w1e+b1),
// GEMM with W2, +b2, silu, attention gate; coalesced async-from-LDS writeback.
// ---------------------------------------------------------------------------
__global__ __launch_bounds__(256)
void edge_mlp_kernel(const float* __restrict__ Hh, const float* __restrict__ Hc,
                     const int* __restrict__ rows, const int* __restrict__ cols,
                     const float* __restrict__ radial, const float* __restrict__ eattr,
                     const float* __restrict__ w1r, const float* __restrict__ w1e,
                     const float* __restrict__ b1,
                     const half_t* __restrict__ W2T, const float* __restrict__ b2,
                     const float* __restrict__ attW, const float* __restrict__ attb,
                     half_t* __restrict__ mOut) {
  __shared__ half_t smem[2 * A_SZ + 2 * B_SZ];   // A bufs | B bufs; aliased as out tile
  __shared__ int   eR[64], eC[64];
  __shared__ float eRad[64], eEA[64];
  __shared__ float red[64];

  int  tid = threadIdx.x;
  long e0  = (long)blockIdx.x * 64;
  if (tid < 64) {
    eR[tid]   = rows[e0 + tid];
    eC[tid]   = cols[e0 + tid];
    eRad[tid] = radial[e0 + tid];
    eEA[tid]  = eattr[e0 + tid];
  }
  __syncthreads();
  int lane = tid & 31, wave = tid >> 5, mgrp = wave >> 1, ngrp = wave & 1;
  int am  = tid >> 2;             // A-builder: 4 threads per row, 8 k each
  int akq = (tid & 3) * 8;
  int   gR = eR[am], gC = eC[am];
  float rd = eRad[am], eaV = eEA[am];

  auto buildA = [&](half_t* dst, int kk) {   // fused gather + rank-1 + bias + silu
    const v4f* hh = (const v4f*)(Hh + (long)gR * 256 + kk + akq);
    const v4f* hc = (const v4f*)(Hc + (long)gC * 256 + kk + akq);
    const v4f* wr = (const v4f*)(w1r + kk + akq);
    const v4f* we = (const v4f*)(w1e + kk + akq);
    const v4f* bb = (const v4f*)(b1 + kk + akq);
    v8h out;
#pragma unroll
    for (int j = 0; j < 2; j++) {
      v4f a = hh[j], b = hc[j], w0 = wr[j], w1 = we[j], bv = bb[j];
#pragma unroll
      for (int c4 = 0; c4 < 4; c4++) {
        float v = a[c4] + b[c4] + rd * w0[c4] + eaV * w1[c4] + bv[c4];
        out[j * 4 + c4] = (half_t)siluf(v);
      }
    }
    *(v8h*)(dst + am * AS + akq) = out;
  };

  v8f acc[8];
#pragma unroll
  for (int t = 0; t < 8; t++)
#pragma unroll
    for (int r = 0; r < 8; r++) acc[t][r] = 0.0f;

  buildA(smem, 0);                               // A buffer 0
  stage_wtile(smem + 2 * A_SZ, W2T, 256, 0, tid); // B buffer 0
  wait_async();
  __syncthreads();

  int cur = 0;
  for (int kk = 0; kk < 256; kk += 32) {
    int nxt = cur ^ 1;
    if (kk + 32 < 256) {
      buildA(smem + nxt * A_SZ, kk + 32);
      stage_wtile(smem + 2 * A_SZ + nxt * B_SZ, W2T, 256, kk + 32, tid);
    }
    const half_t* aCur = smem + cur * A_SZ;
    const half_t* bCur = smem + 2 * A_SZ + cur * B_SZ;
    v16h af = load_afrag(aCur, mgrp, lane);
#pragma unroll
    for (int t = 0; t < 8; t++) {
      v16h bf = load_bfrag(bCur, ngrp * 128 + t * 16, lane);
      acc[t]  = wmma16(af, bf, acc[t]);
    }
    wait_async();
    __syncthreads();
    cur = nxt;
  }

  if (tid < 64) red[tid] = 0.0f;
  __syncthreads();
  float part[8];
#pragma unroll
  for (int r = 0; r < 8; r++) part[r] = 0.0f;
#pragma unroll
  for (int t = 0; t < 8; t++) {
    int col = ngrp * 128 + t * 16 + (lane & 15);
    float bv = b2[col], aw = attW[col];
#pragma unroll
    for (int r = 0; r < 8; r++) {
      float v   = siluf(acc[t][r] + bv);
      acc[t][r] = v;
      part[r]  += v * aw;
    }
  }
#pragma unroll
  for (int r = 0; r < 8; r++) {   // reduce across the 16-lane column group
#pragma unroll
    for (int off = 8; off > 0; off >>= 1) part[r] += __shfl_xor(part[r], off);
  }
  if ((lane & 15) == 0) {
    int rb = mgrp * 16 + ((lane >> 4) & 1) * 8;
#pragma unroll
    for (int r = 0; r < 8; r++) atomicAdd(&red[rb + r], part[r]);
  }
  __syncthreads();
  float   ab   = attb[0];
  half_t* outT = smem;                 // alias: tile buffers are dead now
#pragma unroll
  for (int r = 0; r < 8; r++) {        // att once per row, stage tile in LDS
    int   rowl = mgrp * 16 + ((lane >> 4) & 1) * 8 + r;
    float att  = sigmf(red[rowl] + ab);
#pragma unroll
    for (int t = 0; t < 8; t++) {
      int col = ngrp * 128 + t * 16 + (lane & 15);
      outT[rowl * 256 + col] = (half_t)(acc[t][r] * att);
    }
  }
  __syncthreads();
#pragma unroll
  for (int i = 0; i < 8; i++) {        // coalesced async LDS->global writeback
    int idx = tid + i * 256;           // 2048 x b128 = 64 rows x 256 halves
    int row = idx >> 5;
    int kq  = (idx & 31) * 8;
    async_store_b128(mOut + (e0 + row) * 256 + kq, outT + row * 256 + kq);
  }
  // s_endpgm waits all counters; async stores complete before wave retires
}

// ---------------------------------------------------------------------------
// Fused coord MLP: p = silu(m @ cW1 + cb1); s[e] = p . cW2  (scalar per edge)
// ---------------------------------------------------------------------------
__global__ __launch_bounds__(256)
void coord_mlp_kernel(const half_t* __restrict__ M16, const half_t* __restrict__ WT,
                      const float* __restrict__ b, const float* __restrict__ k2,
                      float* __restrict__ sOut) {
  __shared__ half_t aT[2][A_SZ];
  __shared__ half_t bTt[2][B_SZ];
  __shared__ float red[64];
  int  tid  = threadIdx.x;
  int  lane = tid & 31, wave = tid >> 5, mgrp = wave >> 1, ngrp = wave & 1;
  long e0   = (long)blockIdx.x * 64;

  v8f acc[8];
#pragma unroll
  for (int t = 0; t < 8; t++)
#pragma unroll
    for (int r = 0; r < 8; r++) acc[t][r] = 0.0f;

  stage_atile(aT[0], M16, e0, 256, 0, tid);
  stage_wtile(bTt[0], WT, 256, 0, tid);
  wait_async();
  __syncthreads();

  int cur = 0;
  for (int kk = 0; kk < 256; kk += 32) {
    int nxt = cur ^ 1;
    if (kk + 32 < 256) {
      stage_atile(aT[nxt], M16, e0, 256, kk + 32, tid);
      stage_wtile(bTt[nxt], WT, 256, kk + 32, tid);
    }
    v16h af = load_afrag(aT[cur], mgrp, lane);
#pragma unroll
    for (int t = 0; t < 8; t++) {
      v16h bf = load_bfrag(bTt[cur], ngrp * 128 + t * 16, lane);
      acc[t]  = wmma16(af, bf, acc[t]);
    }
    wait_async();
    __syncthreads();
    cur = nxt;
  }

  if (tid < 64) red[tid] = 0.0f;
  __syncthreads();
  float part[8];
#pragma unroll
  for (int r = 0; r < 8; r++) part[r] = 0.0f;
#pragma unroll
  for (int t = 0; t < 8; t++) {
    int col = ngrp * 128 + t * 16 + (lane & 15);
    float bv = b[col], kv = k2[col];
#pragma unroll
    for (int r = 0; r < 8; r++) part[r] += siluf(acc[t][r] + bv) * kv;
  }
#pragma unroll
  for (int r = 0; r < 8; r++) {
#pragma unroll
    for (int off = 8; off > 0; off >>= 1) part[r] += __shfl_xor(part[r], off);
  }
  if ((lane & 15) == 0) {
    int rb = mgrp * 16 + ((lane >> 4) & 1) * 8;
#pragma unroll
    for (int r = 0; r < 8; r++) atomicAdd(&red[rb + r], part[r]);
  }
  __syncthreads();
  if (tid < 64) sOut[e0 + tid] = red[tid];
}

// ---------------------------------------------------------------------------
// Convert f32 [K,256] matrices to f16 transposed [256,K] (k-contiguous rows).
__global__ void cvtT_kernel(const float* __restrict__ src, half_t* __restrict__ dst,
                            int K, int nmat, long srcMatStride) {
  long total  = (long)nmat * K * 256;
  long stride = (long)gridDim.x * 256;
  for (long i = (long)blockIdx.x * 256 + threadIdx.x; i < total; i += stride) {
    long mat = i / ((long)K * 256);
    long rem = i - mat * (long)K * 256;
    int  k   = (int)(rem >> 8);
    int  n   = (int)(rem & 255);
    dst[mat * (long)K * 256 + (long)n * K + k] = (half_t)src[mat * srcMatStride + rem];
  }
}

__global__ __launch_bounds__(256)
void prep_kernel(const float* __restrict__ xt, const float* __restrict__ t,
                 const float* __restrict__ embW, const float* __restrict__ embB,
                 float* __restrict__ x, float* __restrict__ h, half_t* __restrict__ h16) {
  int n = blockIdx.x, j = threadIdx.x;
  int b = n / PP;
  __shared__ float temb[32];
  float tb = t[b];
  if (j < 16) {
    float lt = logf(tb) * 0.25f;
    float fr = __expf(-(logf(10000.0f) / 15.0f) * (float)j);
    float a  = lt * fr;
    temb[j]      = __sinf(a);
    temb[j + 16] = __cosf(a);
  }
  __syncthreads();
  float acc = embB[j];
#pragma unroll
  for (int i = 0; i < 32; i++) acc += temb[i] * embW[i * 256 + j];
  h[(long)n * 256 + j]   = acc;
  h16[(long)n * 256 + j] = (half_t)acc;
  if (j < 3) {
    float cin = rsqrtf(tb * tb + 1.0f);
    x[n * 3 + j] = xt[n * 3 + j] * cin;
  }
}

__global__ void edge_geom_kernel(const float* __restrict__ x, const int* __restrict__ rows,
                                 const int* __restrict__ cols, float* __restrict__ radial,
                                 float* __restrict__ diffn, float* __restrict__ eattr,
                                 int storeEA) {
  long e = (long)blockIdx.x * 256 + threadIdx.x;
  if (e >= NEDGE) return;
  int r = rows[e], c = cols[e];
  float dx = x[r * 3 + 0] - x[c * 3 + 0];
  float dy = x[r * 3 + 1] - x[c * 3 + 1];
  float dz = x[r * 3 + 2] - x[c * 3 + 2];
  float rad = dx * dx + dy * dy + dz * dz;
  radial[e] = rad;
  float inv = 1.0f / (sqrtf(rad + 1e-8f) + 1.0f);   // NORM_CONST = 1
  diffn[e * 3 + 0] = dx * inv;
  diffn[e * 3 + 1] = dy * inv;
  diffn[e * 3 + 2] = dz * inv;
  if (storeEA) eattr[e] = rad;
}

// per node: aggm = sum of 54 contiguous edge rows of m; x += sum diffn*s; build hn16
__global__ __launch_bounds__(128)
void aggregate_kernel(const half_t* __restrict__ M16, const float* __restrict__ sE,
                      const float* __restrict__ diffn, const half_t* __restrict__ h16,
                      float* __restrict__ x, half_t* __restrict__ hn16) {
  int n = blockIdx.x, j = threadIdx.x;   // j covers 2 columns
  int b = n / PP, i = n % PP;
  long e0 = (long)b * EPB + (long)i * DEG;
  int  j2 = j * 2;
  float a0 = 0.0f, a1 = 0.0f;
  for (int k = 0; k < DEG; k++) {
    v2h p = *(const v2h*)(M16 + (e0 + k) * 256 + j2);
    a0 += (float)p[0];
    a1 += (float)p[1];
  }
  v2h outp; outp[0] = (half_t)a0; outp[1] = (half_t)a1;
  *(v2h*)(hn16 + (long)n * 512 + 256 + j2) = outp;
  *(v2h*)(hn16 + (long)n * 512 + j2) = *(const v2h*)(h16 + (long)n * 256 + j2);
  if (j < 3) {
    float a = 0.0f;
    for (int k = 0; k < DEG; k++) a += diffn[(e0 + k) * 3 + j] * sE[e0 + k];
    x[n * 3 + j] += a;
  }
}

__global__ void finalize_kernel(const float* __restrict__ xt, const float* __restrict__ t,
                                const float* __restrict__ x, float* __restrict__ out) {
  int b = blockIdx.x, tid = threadIdx.x;
  __shared__ float mean[3];
  if (tid < 3) {
    float s = 0.0f;
    for (int p = 0; p < PP; p++) s += x[(b * PP + p) * 3 + tid];
    mean[tid] = s / (float)PP;
  }
  __syncthreads();
  float tb = t[b];
  float cs = 1.0f / (tb * tb + 1.0f);
  float co = tb * rsqrtf(1.0f + tb * tb);
  for (int idx = tid; idx < PP * 3; idx += 64) {
    int p = idx / 3, d = idx % 3;
    long g = (long)(b * PP + p) * 3 + d;
    out[g] = xt[g] * cs + (x[g] - mean[d]) * co;
  }
}

// ---------------------------------------------------------------------------
extern "C" void kernel_launch(void* const* d_in, const int* in_sizes, int n_in,
                              void* d_out, int out_size, void* d_ws, size_t ws_size,
                              hipStream_t stream) {
  (void)in_sizes; (void)n_in; (void)out_size; (void)ws_size;
  const float* xt   = (const float*)d_in[0];
  const float* tt   = (const float*)d_in[1];
  const int*   rows = (const int*)d_in[2];
  const int*   cols = (const int*)d_in[3];
  const float* embW = (const float*)d_in[4];
  const float* embB = (const float*)d_in[5];
  const float* eW1  = (const float*)d_in[6];
  const float* eb1  = (const float*)d_in[7];
  const float* eW2  = (const float*)d_in[8];
  const float* eb2  = (const float*)d_in[9];
  const float* attW = (const float*)d_in[10];
  const float* attb = (const float*)d_in[11];
  const float* nW1  = (const float*)d_in[12];
  const float* nb1  = (const float*)d_in[13];
  const float* nW2  = (const float*)d_in[14];
  const float* nb2  = (const float*)d_in[15];
  const float* cW1  = (const float*)d_in[16];
  const float* cb1  = (const float*)d_in[17];
  const float* cW2  = (const float*)d_in[18];

  char*  base = (char*)d_ws;
  size_t off  = 0;
  auto alloc = [&](size_t bytes) -> char* {
    char* p = base + off;
    off = (off + bytes + 255) & ~(size_t)255;
    return p;
  };
  float*  x     = (float*)alloc((size_t)NNODE * 3 * 4);
  float*  h     = (float*)alloc((size_t)NNODE * 256 * 4);
  half_t* h16   = (half_t*)alloc((size_t)NNODE * 256 * 2);
  half_t* hn16  = (half_t*)alloc((size_t)NNODE * 512 * 2);
  half_t* t16   = (half_t*)alloc((size_t)NNODE * 256 * 2);
  float*  Hh    = (float*)alloc((size_t)NNODE * 256 * 4);
  float*  Hc    = (float*)alloc((size_t)NNODE * 256 * 4);
  float*  ea    = (float*)alloc((size_t)NEDGE * 4);
  float*  rad   = (float*)alloc((size_t)NEDGE * 4);
  float*  dfn   = (float*)alloc((size_t)NEDGE * 3 * 4);
  float*  sE    = (float*)alloc((size_t)NEDGE * 4);
  half_t* m16   = (half_t*)alloc((size_t)NEDGE * 256 * 2);
  half_t* eW1hT = (half_t*)alloc((size_t)2 * NLAY * 65536 * 2); // [top L][bot L]
  half_t* eW2hT = (half_t*)alloc((size_t)NLAY * 65536 * 2);
  half_t* nW1hT = (half_t*)alloc((size_t)NLAY * 512 * 256 * 2);
  half_t* nW2hT = (half_t*)alloc((size_t)NLAY * 65536 * 2);
  half_t* cW1hT = (half_t*)alloc((size_t)NLAY * 65536 * 2);

  // f32 -> f16 + transpose to [256,K] (k-contiguous) once per call
  cvtT_kernel<<<512, 256, 0, stream>>>(eW1, eW1hT, 256, NLAY, (long)514 * 256);
  cvtT_kernel<<<512, 256, 0, stream>>>(eW1 + 65536, eW1hT + (size_t)NLAY * 65536,
                                       256, NLAY, (long)514 * 256);
  cvtT_kernel<<<512, 256, 0, stream>>>(eW2, eW2hT, 256, NLAY, 65536);
  cvtT_kernel<<<512, 256, 0, stream>>>(nW1, nW1hT, 512, NLAY, (long)512 * 256);
  cvtT_kernel<<<512, 256, 0, stream>>>(nW2, nW2hT, 256, NLAY, 65536);
  cvtT_kernel<<<512, 256, 0, stream>>>(cW1, cW1hT, 256, NLAY, 65536);

  prep_kernel<<<NNODE, 256, 0, stream>>>(xt, tt, embW, embB, x, h, h16);

  for (int l = 0; l < NLAY; l++) {
    const size_t w1o = (size_t)l * 514 * 256;
    edge_geom_kernel<<<(NEDGE + 255) / 256, 256, 0, stream>>>(x, rows, cols, rad, dfn, ea,
                                                              l == 0 ? 1 : 0);
    // Hh = h @ W1[0:256,:], Hc = h @ W1[256:512,:]   (factored edge GEMM1)
    gemm_kernel<8><<<NNODE / 64, 256, 0, stream>>>(h16, 256, 256,
                                                   eW1hT + (size_t)l * 65536,
                                                   nullptr, Hh, nullptr);
    gemm_kernel<8><<<NNODE / 64, 256, 0, stream>>>(h16, 256, 256,
                                                   eW1hT + (size_t)(NLAY + l) * 65536,
                                                   nullptr, Hc, nullptr);
    edge_mlp_kernel<<<NEDGE / 64, 256, 0, stream>>>(
        Hh, Hc, rows, cols, rad, ea,
        eW1 + w1o + 512 * 256, eW1 + w1o + 513 * 256, eb1 + l * 256,
        eW2hT + (size_t)l * 65536, eb2 + l * 256, attW + l * 256, attb + l, m16);
    coord_mlp_kernel<<<NEDGE / 64, 256, 0, stream>>>(
        m16, cW1hT + (size_t)l * 65536, cb1 + l * 256, cW2 + l * 256, sE);
    aggregate_kernel<<<NNODE, 128, 0, stream>>>(m16, sE, dfn, h16, x, hn16);
    // node MLP: t = silu(hn @ n1 + b); h += t @ n2 + b
    gemm_kernel<1 | 4><<<NNODE / 64, 256, 0, stream>>>(hn16, 512, 512,
                                                       nW1hT + (size_t)l * 512 * 256,
                                                       nb1 + l * 256, nullptr, t16);
    gemm_kernel<2 | 4><<<NNODE / 64, 256, 0, stream>>>(t16, 256, 256,
                                                       nW2hT + (size_t)l * 65536,
                                                       nb2 + l * 256, h, h16);
  }

  finalize_kernel<<<PB, 64, 0, stream>>>(xt, tt, x, (float*)d_out);
}
